// LearnableUpsampling_14233521619326
// MI455X (gfx1250) — matrix operand order repaired and checked by reading, not compiled, not run on hardware
//
#include <hip/hip_runtime.h>
#include <hip/hip_bf16.h>
#include <stdint.h>

// ---------------------------------------------------------------------------
// LearnableUpsampling for MI455X (gfx1250, wave32, WMMA).
// Heavy GEMMs run on v_wmma_f32_16x16x32_bf16 (f32 accumulate); linear_w is
// algebraically folded into the bqtk,bkh einsum via U = V @ lw_q^T so the
// per-tile GEMM is a single K=768 WMMA chain reading A from LDS.
// Bulk global->LDS tile copies use gfx1250 async-LDS loads (ASYNCcnt path).
// ---------------------------------------------------------------------------

#define Dh    192   // D
#define COc   8     // CO
#define DWq   4     // DW
#define DCp   2     // DC
#define Bb    16    // B
#define SRCk  192   // SRC (== K of the attention)
#define Tt    1000  // MAX_SEQ_LEN
#define NTILE 63    // ceil(Tt/16)

typedef __attribute__((ext_vector_type(16))) __bf16 v16bf;
typedef __attribute__((ext_vector_type(8)))  float  v8f;

static __device__ __forceinline__ float siluf(float x) { return x / (1.f + __expf(-x)); }

static __device__ __forceinline__ v8f wmma_bf16(v16bf a, v16bf b, v8f c) {
  // (neg_a, A, neg_b, B, c_mod, C, reuse_a, reuse_b)
  return __builtin_amdgcn_wmma_f32_16x16x32_bf16(false, a, false, b, (short)0, c, false, false);
}

// A-fragment element e of 16-bit 16x32 A:  K = (lane/16)*16 + e
static __device__ __forceinline__ int a_koff(int e, int half) {
  return (half << 4) + e;
}

// Per-lane async global->LDS copy of 16 bytes (GLOBAL_LOAD_ASYNC_TO_LDS_B128,
// GV addressing, tracked on ASYNCcnt). ldsAddr = wave-relative LDS byte addr
// (low 32 bits of the generic shared pointer, per the aperture mapping).
static __device__ __forceinline__ void async_ld_b128(unsigned ldsAddr, const void* gaddr) {
  asm volatile("global_load_async_to_lds_b128 %0, %1, off"
               :: "v"(ldsAddr), "v"((unsigned long long)(uintptr_t)gaddr)
               : "memory");
}
static __device__ __forceinline__ void wait_async0() {
  asm volatile("s_wait_asynccnt 0" ::: "memory");
}

// ---------------------------------------------------------------------------
// Kernel 1: duration scan + conv blocks (both conv_w and conv_c).
// ---------------------------------------------------------------------------
__global__ __launch_bounds__(192) void prep_kernel(
    const float* __restrict__ dur, const float* __restrict__ V,
    const float* __restrict__ cw_w, const float* __restrict__ cw_b,
    const float* __restrict__ cw_bng, const float* __restrict__ cw_bnb,
    const float* __restrict__ cw_bnm, const float* __restrict__ cw_bnv,
    const float* __restrict__ cw_lng, const float* __restrict__ cw_lnb,
    const float* __restrict__ cc_w, const float* __restrict__ cc_b,
    const float* __restrict__ cc_bng, const float* __restrict__ cc_bnb,
    const float* __restrict__ cc_bnm, const float* __restrict__ cc_bnv,
    const float* __restrict__ cc_lng, const float* __restrict__ cc_lnb,
    float* __restrict__ ws_ek, float* __restrict__ ws_sk, int* __restrict__ ws_mel,
    float* __restrict__ ws_Vw, float* __restrict__ ws_Vc,
    float* __restrict__ out_mask, float* __restrict__ out_mellen)
{
  const int b = blockIdx.x;
  const int tid = threadIdx.x;

  __shared__ float s_scan[SRCk];
  __shared__ float s_w[2 * COc * Dh * 3];   // conv_w then conv_c weights (36 KB)

  for (int i = tid; i < COc * Dh * 3; i += 192) {
    s_w[i] = cw_w[i];
    s_w[COc * Dh * 3 + i] = cc_w[i];
  }

  const float d0 = dur[b * SRCk + tid];
  s_scan[tid] = d0;
  __syncthreads();
  // Hillis-Steele inclusive scan over 192 entries.
  for (int off = 1; off < SRCk; off <<= 1) {
    float v = s_scan[tid];
    float add = (tid >= off) ? s_scan[tid - off] : 0.f;
    __syncthreads();
    s_scan[tid] = v + add;
    __syncthreads();
  }
  const float ekv = s_scan[tid];
  ws_ek[b * SRCk + tid] = ekv;
  ws_sk[b * SRCk + tid] = ekv - d0;

  int mel = (int)rintf(s_scan[SRCk - 1]);
  mel = mel < 0 ? 0 : (mel > Tt ? Tt : mel);
  if (tid == 0) { ws_mel[b] = mel; out_mellen[b] = (float)mel; }
  for (int t = tid; t < Tt; t += 192) out_mask[b * Tt + t] = (t >= mel) ? 1.f : 0.f;

  // 1D conv over source positions, SAME pad, KS=3 (cross-correlation).
  const int s = tid;
  float accw[COc], accc[COc];
#pragma unroll
  for (int c = 0; c < COc; ++c) { accw[c] = 0.f; accc[c] = 0.f; }
  const float* Vb = V + (size_t)b * SRCk * Dh;
  for (int d = 0; d < Dh; ++d) {
    float v0 = (s > 0)        ? Vb[(size_t)(s - 1) * Dh + d] : 0.f;
    float v1 =                  Vb[(size_t)s * Dh + d];
    float v2 = (s < SRCk - 1) ? Vb[(size_t)(s + 1) * Dh + d] : 0.f;
#pragma unroll
    for (int c = 0; c < COc; ++c) {
      const float* ww = &s_w[(c * Dh + d) * 3];
      accw[c] += v0 * ww[0] + v1 * ww[1] + v2 * ww[2];
      const float* wc = &s_w[COc * Dh * 3 + (c * Dh + d) * 3];
      accc[c] += v0 * wc[0] + v1 * wc[1] + v2 * wc[2];
    }
  }
  // BN + SiLU + LayerNorm(CO) for both branches.
  {
    float y[COc]; float mn = 0.f;
#pragma unroll
    for (int c = 0; c < COc; ++c) {
      float x = (accw[c] + cw_b[c] - cw_bnm[c]) * rsqrtf(cw_bnv[c] + 1e-5f) * cw_bng[c] + cw_bnb[c];
      y[c] = siluf(x); mn += y[c];
    }
    mn *= (1.f / COc);
    float vr = 0.f;
#pragma unroll
    for (int c = 0; c < COc; ++c) { float df = y[c] - mn; vr += df * df; }
    float rs = rsqrtf(vr * (1.f / COc) + 1e-5f);
#pragma unroll
    for (int c = 0; c < COc; ++c)
      ws_Vw[((size_t)b * SRCk + s) * COc + c] = (y[c] - mn) * rs * cw_lng[c] + cw_lnb[c];
  }
  {
    float y[COc]; float mn = 0.f;
#pragma unroll
    for (int c = 0; c < COc; ++c) {
      float x = (accc[c] + cc_b[c] - cc_bnm[c]) * rsqrtf(cc_bnv[c] + 1e-5f) * cc_bng[c] + cc_bnb[c];
      y[c] = siluf(x); mn += y[c];
    }
    mn *= (1.f / COc);
    float vr = 0.f;
#pragma unroll
    for (int c = 0; c < COc; ++c) { float df = y[c] - mn; vr += df * df; }
    float rs = rsqrtf(vr * (1.f / COc) + 1e-5f);
#pragma unroll
    for (int c = 0; c < COc; ++c)
      ws_Vc[((size_t)b * SRCk + s) * COc + c] = (y[c] - mn) * rs * cc_lng[c] + cc_lnb[c];
  }
}

// ---------------------------------------------------------------------------
// Kernel 2: U_t[b][q][d][k] = sum_h V[b,k,h] * lw[d, q*D+h], stored as bf16,
// k-contiguous so the main GEMM's B-fragments are one contiguous 32B load.
// One wave per 16x16 tile: M=d, N=k, K=h (192 -> 6 WMMA).
// ---------------------------------------------------------------------------
__global__ __launch_bounds__(32) void u_kernel(
    const float* __restrict__ V, const float* __restrict__ lw,
    __bf16* __restrict__ Ut)
{
  const int tile = blockIdx.x;           // 0..143
  const int q = blockIdx.y, b = blockIdx.z;
  const int mt = tile / 12, nt = tile % 12;   // mt: d-tile, nt: k-tile
  const int lane = threadIdx.x, half = lane >> 4, lm = lane & 15;

  const float* Arow = lw + (size_t)(mt * 16 + lm) * (DWq * Dh) + q * Dh; // lw[d][q*192 + h]
  const float* Brow = V + ((size_t)b * SRCk + nt * 16 + lm) * Dh;        // V[b][k][h]

  v8f acc = {};
  for (int h0 = 0; h0 < Dh; h0 += 32) {
    v16bf a, bv;
#pragma unroll
    for (int e = 0; e < 16; ++e) {
      a[e]  = (__bf16)Arow[h0 + a_koff(e, half)];
      bv[e] = (__bf16)Brow[h0 + (half << 4) + e];
    }
    acc = wmma_bf16(a, bv, acc);
  }
  __bf16* Uq = Ut + ((size_t)(b * DWq + q) * Dh) * Dh;
#pragma unroll
  for (int i = 0; i < 8; ++i) {
    int dd = mt * 16 + i + 8 * half;               // D-frag: M = i + 8*(lane/16)
    Uq[(size_t)dd * Dh + nt * 16 + lm] = (__bf16)acc[i];
  }
}

// ---------------------------------------------------------------------------
// Kernel 3: one (b, 16-row t-tile) per block; 192 threads = 6 waves.
// ---------------------------------------------------------------------------
__global__ __launch_bounds__(192) void main_kernel(
    const int* __restrict__ src_lens,
    const float* __restrict__ ws_ek, const float* __restrict__ ws_sk,
    const int* __restrict__ ws_mel,
    const float* __restrict__ ws_Vw, const float* __restrict__ ws_Vc,
    const __bf16* __restrict__ Ut,
    const float* __restrict__ sw_w1, const float* __restrict__ sw_b1,
    const float* __restrict__ sw_w2, const float* __restrict__ sw_b2,
    const float* __restrict__ sc_w1, const float* __restrict__ sc_b1,
    const float* __restrict__ sc_w2, const float* __restrict__ sc_b2,
    const float* __restrict__ lw_b,
    const float* __restrict__ le_w, const float* __restrict__ le_b,
    const float* __restrict__ ln_g, const float* __restrict__ ln_b,
    const float* __restrict__ po_w, const float* __restrict__ po_b,
    float* __restrict__ out_rep, float* __restrict__ out_wp)
{
  const int tile = blockIdx.x, b = blockIdx.y;
  const int tid = threadIdx.x;
  const int t0 = tile * 16;

  __shared__ float s_ek[SRCk], s_sk[SRCk];
  __shared__ __attribute__((aligned(16))) float s_vwc[SRCk * COc];  // Vw, later Vc (6 KB)
  __shared__ float s_buf[16 * SRCk * DWq];     // Wsc/Wm, later reused as rep (48 KB)
  __shared__ float s_mx[64], s_sum[64];
  __shared__ float s_red[16 * 12 * 8];         // deterministic o2 partials (6 KB)
  __shared__ float s_o2[16 * 8];
  __shared__ float s_mean[16], s_rstd[16];
  __shared__ float s_sw[64];                   // w1(40)|b1(4)|w2(16)|b2(4)
  __shared__ float s_sc[28];                   // w1(20)|b1(2)|w2(4)|b2(2)
  __shared__ int   s_mel;

  // Kick off the async Vw tile copy first so it overlaps the scalar staging.
  const unsigned ldsVwc = (unsigned)(uintptr_t)(&s_vwc[0]);
  {
    const char* g = (const char*)(ws_Vw + (size_t)b * SRCk * COc);
#pragma unroll
    for (int it = 0; it < 2; ++it) {            // 2 x 192 lanes x 16B = 6144B
      const unsigned off = (unsigned)(it * 192 + tid) * 16u;
      async_ld_b128(ldsVwc + off, g + off);
    }
  }

  if (tid == 0) s_mel = ws_mel[b];
  s_ek[tid] = ws_ek[b * SRCk + tid];
  s_sk[tid] = ws_sk[b * SRCk + tid];
  if      (tid < 40) s_sw[tid] = sw_w1[tid];
  else if (tid < 44) s_sw[tid] = sw_b1[tid - 40];
  else if (tid < 60) s_sw[tid] = sw_w2[tid - 44];
  else if (tid < 64) s_sw[tid] = sw_b2[tid - 60];
  else if (tid < 84) s_sc[tid - 64] = sc_w1[tid - 64];
  else if (tid < 86) s_sc[tid - 64] = sc_b1[tid - 84];
  else if (tid < 90) s_sc[tid - 64] = sc_w2[tid - 86];
  else if (tid < 92) s_sc[tid - 64] = sc_b2[tid - 90];
  wait_async0();
  __syncthreads();

  const int k = tid;                      // this thread owns source index k
  const int slen = src_lens[b];
  const bool smask = (k >= slen);
  const int mel = s_mel;
  const float skv = s_sk[k], ekv = s_ek[k];

  // ---- Pass A: swish_w -> logits Wsc[t][k][q] ----
  float vwk[COc];
#pragma unroll
  for (int c = 0; c < COc; ++c) vwk[c] = s_vwc[k * COc + c];
  for (int tl = 0; tl < 16; ++tl) {
    const int tg = t0 + tl;
    const float tt = (float)(tg + 1);
    const bool amask = smask || (tg >= mel);
    const float S = amask ? 0.f : (tt - skv);
    const float E = amask ? 0.f : (ekv - tt);
    float h1[DWq];
#pragma unroll
    for (int o = 0; o < DWq; ++o) {
      float acc = s_sw[40 + o] + s_sw[o * 10 + 0] * S + s_sw[o * 10 + 1] * E;
#pragma unroll
      for (int c = 0; c < COc; ++c) acc += s_sw[o * 10 + 2 + c] * vwk[c];
      h1[o] = siluf(acc);
    }
#pragma unroll
    for (int o = 0; o < DWq; ++o) {
      float acc = s_sw[60 + o];
#pragma unroll
      for (int i2 = 0; i2 < DWq; ++i2) acc += s_sw[44 + o * 4 + i2] * h1[i2];
      float w = siluf(acc);
      s_buf[(tl * SRCk + k) * DWq + o] = smask ? -1e30f : w;
    }
  }
  __syncthreads();

  // ---- Softmax stats over k, per (t,q): 64 threads, fixed order ----
  if (tid < 64) {
    const int tl = tid >> 2, q = tid & 3;
    float mx = -1e30f;
    for (int kk = 0; kk < SRCk; ++kk) mx = fmaxf(mx, s_buf[(tl * SRCk + kk) * DWq + q]);
    float sm = 0.f;
    for (int kk = 0; kk < SRCk; ++kk) sm += __expf(s_buf[(tl * SRCk + kk) * DWq + q] - mx);
    s_mx[tid] = mx;
    s_sum[tid] = (sm > 0.f) ? 1.f / sm : 0.f;
  }
  __syncthreads();

  // ---- Normalize, apply mel mask, write Wp[b,q,t,k] ----
  for (int tl = 0; tl < 16; ++tl) {
    const int tg = t0 + tl;
    const bool mm = (tg >= mel);
#pragma unroll
    for (int q = 0; q < DWq; ++q) {
      float x = s_buf[(tl * SRCk + k) * DWq + q];
      float w = __expf(x - s_mx[tl * 4 + q]) * s_sum[tl * 4 + q];
      if (smask || mm) w = 0.f;
      s_buf[(tl * SRCk + k) * DWq + q] = w;
      if (tg < Tt)
        out_wp[(((size_t)(b * DWq + q)) * Tt + tg) * SRCk + k] = w;
    }
  }
  __syncthreads();

  // Async-swap the Vc tile into the same LDS buffer.
  {
    const char* g = (const char*)(ws_Vc + (size_t)b * SRCk * COc);
#pragma unroll
    for (int it = 0; it < 2; ++it) {
      const unsigned off = (unsigned)(it * 192 + tid) * 16u;
      async_ld_b128(ldsVwc + off, g + off);
    }
    wait_async0();
  }
  __syncthreads();

  // ---- Pass B: swish_c -> C, deterministic o2[t][q*2+p] reduction ----
  {
    const int tt_row = tid / 12;         // 0..15
    const int l12 = tid % 12;
    float part[8];
#pragma unroll
    for (int j = 0; j < 8; ++j) part[j] = 0.f;
    const int tg = t0 + tt_row;
    const float ttf = (float)(tg + 1);
    for (int kk = l12; kk < SRCk; kk += 12) {
      const bool smk = (kk >= slen);
      const bool amask = smk || (tg >= mel);
      const float S = amask ? 0.f : (ttf - s_sk[kk]);
      const float E = amask ? 0.f : (s_ek[kk] - ttf);
      float h1[DCp];
#pragma unroll
      for (int o = 0; o < DCp; ++o) {
        float acc = s_sc[20 + o] + s_sc[o * 10 + 0] * S + s_sc[o * 10 + 1] * E;
#pragma unroll
        for (int c = 0; c < COc; ++c) acc += s_sc[o * 10 + 2 + c] * s_vwc[kk * COc + c];
        h1[o] = siluf(acc);
      }
      const float C0 = siluf(s_sc[26] + s_sc[22] * h1[0] + s_sc[23] * h1[1]);
      const float C1 = siluf(s_sc[27] + s_sc[24] * h1[0] + s_sc[25] * h1[1]);
#pragma unroll
      for (int q = 0; q < DWq; ++q) {
        const float w = s_buf[(tt_row * SRCk + kk) * DWq + q];
        part[q * 2 + 0] += w * C0;
        part[q * 2 + 1] += w * C1;
      }
    }
#pragma unroll
    for (int j = 0; j < 8; ++j) s_red[(tt_row * 12 + l12) * 8 + j] = part[j];
  }
  __syncthreads();
  if (tid < 128) {
    const int tl = tid >> 3, j = tid & 7;
    float acc = 0.f;
    for (int r = 0; r < 12; ++r) acc += s_red[(tl * 12 + r) * 8 + j];
    s_o2[tid] = acc;
  }
  __syncthreads();

  // ---- o1 GEMM: (16 x 768) x (768 x 192) via WMMA bf16, A from LDS ----
  const int wave = tid >> 5, lane = tid & 31, half = lane >> 4, lm = lane & 15;
  v8f acc0 = {}, acc1 = {};
  {
    const int n0 = wave * 2, n1 = wave * 2 + 1;
    for (int q = 0; q < DWq; ++q) {
      const __bf16* Uq = Ut + ((size_t)(b * DWq + q) * Dh) * Dh;
      const __bf16* Ub0 = Uq + (size_t)(n0 * 16 + lm) * Dh;
      const __bf16* Ub1 = Uq + (size_t)(n1 * 16 + lm) * Dh;
      if (q < DWq - 1) {  // warm the next U panel (L2-resident; global_prefetch_b8)
        const __bf16* Un = Ut + ((size_t)(b * DWq + q + 1) * Dh) * Dh;
        __builtin_prefetch(Un + (size_t)(n0 * 16 + lm) * Dh, 0, 1);
        __builtin_prefetch(Un + (size_t)(n1 * 16 + lm) * Dh, 0, 1);
      }
      for (int k0 = 0; k0 < SRCk; k0 += 32) {
        v16bf a;
#pragma unroll
        for (int e = 0; e < 16; ++e)
          a[e] = (__bf16)s_buf[(lm * SRCk + k0 + a_koff(e, half)) * DWq + q];
        v16bf b0, b1;
        __builtin_memcpy(&b0, Ub0 + k0 + (half << 4), sizeof(v16bf));
        __builtin_memcpy(&b1, Ub1 + k0 + (half << 4), sizeof(v16bf));
        acc0 = wmma_bf16(a, b0, acc0);
        acc1 = wmma_bf16(a, b1, acc1);
      }
    }
  }
  __syncthreads();                 // all Wm reads complete -> safe to reuse s_buf
  float* s_rep = s_buf;            // rep[t][d], 16x192
  {
    const int n0 = wave * 2, n1 = wave * 2 + 1;
#pragma unroll
    for (int i = 0; i < 8; ++i) {
      const int m = i + 8 * half;
      s_rep[m * Dh + n0 * 16 + lm] = acc0[i];
      s_rep[m * Dh + n1 * 16 + lm] = acc1[i];
    }
  }
  __syncthreads();

  // ---- add linear biases + o2 @ linear_einsum ----
  {
    const int d = tid;
    const float lbv = lw_b[d] + le_b[d];
    float wev[8];
#pragma unroll
    for (int j = 0; j < 8; ++j) wev[j] = le_w[d * 8 + j];
    for (int tl = 0; tl < 16; ++tl) {
      float o2l = 0.f;
#pragma unroll
      for (int j = 0; j < 8; ++j) o2l += s_o2[tl * 8 + j] * wev[j];
      s_rep[tl * Dh + d] += lbv + o2l;
    }
  }
  __syncthreads();

  // ---- LayerNorm over D per row ----
  if (tid < 16) {
    float mn = 0.f;
    for (int d = 0; d < Dh; ++d) mn += s_rep[tid * Dh + d];
    mn *= (1.f / Dh);
    float vr = 0.f;
    for (int d = 0; d < Dh; ++d) { float df = s_rep[tid * Dh + d] - mn; vr += df * df; }
    s_mean[tid] = mn;
    s_rstd[tid] = rsqrtf(vr * (1.f / Dh) + 1e-5f);
  }
  __syncthreads();
  {
    const int d = tid;
    const float g = ln_g[d], bb = ln_b[d];
    for (int tl = 0; tl < 16; ++tl) {
      const int tg = t0 + tl;
      float v = (s_rep[tl * Dh + d] - s_mean[tl]) * s_rstd[tl] * g + bb;
      if (tg >= mel) v = 0.f;            // rep masked before proj_o (ref semantics)
      s_rep[tl * Dh + d] = v;
    }
  }
  __syncthreads();

  // ---- proj_o GEMM: (16 x 192) x (192 x 384), 24 n-tiles / 6 waves ----
  {
    v8f pacc[4] = {};
    for (int k0 = 0; k0 < Dh; k0 += 32) {
      v16bf a;
#pragma unroll
      for (int e = 0; e < 16; ++e)
        a[e] = (__bf16)s_rep[lm * Dh + k0 + a_koff(e, half)];
#pragma unroll
      for (int j = 0; j < 4; ++j) {
        const int n = (wave * 4 + j) * 16;
        const float* prow = po_w + (size_t)(n + lm) * Dh + k0 + (half << 4);
        v16bf bv;
#pragma unroll
        for (int e = 0; e < 16; ++e) bv[e] = (__bf16)prow[e];
        pacc[j] = wmma_bf16(a, bv, pacc[j]);
      }
    }
#pragma unroll
    for (int j = 0; j < 4; ++j) {
      const int n = (wave * 4 + j) * 16;
      const float pbv = po_b[n + lm];
#pragma unroll
      for (int i = 0; i < 8; ++i) {
        const int m = i + 8 * half;
        const int tg = t0 + m;
        if (tg < Tt)
          out_rep[((size_t)b * Tt + tg) * (2 * Dh) + n + lm] = pacc[j][i] + pbv;
      }
    }
  }
}

// ---------------------------------------------------------------------------
// Host launcher.
// Input order assumed = setup_inputs() dict insertion order, params flattened
// in their own insertion order:
//   0 duration, 1 V, 2 src_lens, 3 src_mask(bool, unused), 4 max_src_len,
//   5 max_mel_len,
//   6..13  conv_w:  w,b,bn_g,bn_b,bn_m,bn_v,ln_g,ln_b
//   14..21 conv_c:  w,b,bn_g,bn_b,bn_m,bn_v,ln_g,ln_b
//   22..25 swish_w: w1,b1,w2,b2
//   26..29 swish_c: w1,b1,w2,b2
//   30..31 linear_w: w,b      32..33 linear_einsum: w,b
//   34..35 ln: g,b            36..37 proj_o: w,b
// Output: rep(16*1000*384) | mel_mask(16*1000) | mel_len(16) | Wp(16*4*1000*192)
// ---------------------------------------------------------------------------
extern "C" void kernel_launch(void* const* d_in, const int* in_sizes, int n_in,
                              void* d_out, int out_size, void* d_ws, size_t ws_size,
                              hipStream_t stream)
{
  (void)in_sizes; (void)n_in; (void)out_size; (void)ws_size;

  const float* dur      = (const float*)d_in[0];
  const float* V        = (const float*)d_in[1];
  const int*   src_lens = (const int*)d_in[2];

  const float* cw_w   = (const float*)d_in[6];
  const float* cw_b   = (const float*)d_in[7];
  const float* cw_bng = (const float*)d_in[8];
  const float* cw_bnb = (const float*)d_in[9];
  const float* cw_bnm = (const float*)d_in[10];
  const float* cw_bnv = (const float*)d_in[11];
  const float* cw_lng = (const float*)d_in[12];
  const float* cw_lnb = (const float*)d_in[13];
  const float* cc_w   = (const float*)d_in[14];
  const float* cc_b   = (const float*)d_in[15];
  const float* cc_bng = (const float*)d_in[16];
  const float* cc_bnb = (const float*)d_in[17];
  const float* cc_bnm = (const float*)d_in[18];
  const float* cc_bnv = (const float*)d_in[19];
  const float* cc_lng = (const float*)d_in[20];
  const float* cc_lnb = (const float*)d_in[21];
  const float* sw_w1  = (const float*)d_in[22];
  const float* sw_b1  = (const float*)d_in[23];
  const float* sw_w2  = (const float*)d_in[24];
  const float* sw_b2  = (const float*)d_in[25];
  const float* sc_w1  = (const float*)d_in[26];
  const float* sc_b1  = (const float*)d_in[27];
  const float* sc_w2  = (const float*)d_in[28];
  const float* sc_b2  = (const float*)d_in[29];
  const float* lw_w   = (const float*)d_in[30];
  const float* lw_b   = (const float*)d_in[31];
  const float* le_w   = (const float*)d_in[32];
  const float* le_b   = (const float*)d_in[33];
  const float* ln_g   = (const float*)d_in[34];
  const float* ln_b   = (const float*)d_in[35];
  const float* po_w   = (const float*)d_in[36];
  const float* po_b   = (const float*)d_in[37];

  // Workspace carve (~5 MB); keep every segment 16B-aligned for async b128.
  char* p = (char*)d_ws;
  float* ws_ek = (float*)p; p += (size_t)Bb * SRCk * sizeof(float);
  float* ws_sk = (float*)p; p += (size_t)Bb * SRCk * sizeof(float);
  int*   ws_mel = (int*)p;  p += 16 * sizeof(int);
  float* ws_Vw = (float*)p; p += (size_t)Bb * SRCk * COc * sizeof(float);
  float* ws_Vc = (float*)p; p += (size_t)Bb * SRCk * COc * sizeof(float);
  p = (char*)(((uintptr_t)p + 255u) & ~(uintptr_t)255u);
  __bf16* Ut = (__bf16*)p;  // [B][DW][D][K] bf16, 4.72 MB

  // Output segments.
  float* out = (float*)d_out;
  float* out_rep    = out;
  float* out_mask   = out_rep + (size_t)Bb * Tt * 2 * Dh;
  float* out_mellen = out_mask + (size_t)Bb * Tt;
  float* out_wp     = out_mellen + Bb;

  prep_kernel<<<dim3(Bb), dim3(192), 0, stream>>>(
      dur, V,
      cw_w, cw_b, cw_bng, cw_bnb, cw_bnm, cw_bnv, cw_lng, cw_lnb,
      cc_w, cc_b, cc_bng, cc_bnb, cc_bnm, cc_bnv, cc_lng, cc_lnb,
      ws_ek, ws_sk, ws_mel, ws_Vw, ws_Vc, out_mask, out_mellen);

  u_kernel<<<dim3(144, DWq, Bb), dim3(32), 0, stream>>>(V, lw_w, Ut);

  main_kernel<<<dim3(NTILE, Bb), dim3(192), 0, stream>>>(
      src_lens, ws_ek, ws_sk, ws_mel, ws_Vw, ws_Vc, Ut,
      sw_w1, sw_b1, sw_w2, sw_b2, sc_w1, sc_b1, sc_w2, sc_b2,
      lw_b, le_w, le_b, ln_g, ln_b, po_w, po_b,
      out_rep, out_wp);
}
